// FusedMoE_35948876268095
// MI455X (gfx1250) — compile-verified, matrix operand order checked by reading, not run on hardware
//
#include <hip/hip_runtime.h>
#include <stdint.h>

#define DEVINL __device__ __forceinline__

typedef __attribute__((ext_vector_type(16))) __bf16 v16bf;
typedef __attribute__((ext_vector_type(8)))  float  v8f;

constexpr int E_ = 8;
constexpr int H_ = 1024;   // hidden (K of gemm1, N of gemm2)
constexpr int I_ = 2048;   // intermediate
constexpr int T_ = 1024;   // tokens
constexpr int BM = 64, BN = 64, BK = 32;
constexpr int MAX_PAIRS  = 2560;            // 2*T + 8*64 padding, rounded to 64
constexpr int MAX_MTILES = MAX_PAIRS / BM;  // 40
constexpr int LDSS = 40;                    // LDS row stride in bf16 elems (80B, 16B aligned)
constexpr int TILE_E = BM * LDSS;           // elems per LDS tile buffer

// ---------- helpers ----------
DEVINL unsigned bf16_bits_rne(float f) {
  unsigned u = __float_as_uint(f);
  return (u + 0x7FFFu + ((u >> 16) & 1u)) >> 16;   // RNE f32 -> bf16
}
DEVINL unsigned pack2_rne(float lo, float hi) {
  return bf16_bits_rne(lo) | (bf16_bits_rne(hi) << 16);
}
// Truncating pack: exact for MXFP4-dequantized weights (grid values * 2^k are
// exactly representable in bf16), 1 VALU op (v_perm_b32) per 2 elements.
DEVINL unsigned pack2_trunc(float lo, float hi) {
#if __has_builtin(__builtin_amdgcn_perm)
  return __builtin_amdgcn_perm(__float_as_uint(hi), __float_as_uint(lo),
                               0x07060302u);       // single v_perm_b32
#else
  return (__float_as_uint(hi) & 0xFFFF0000u) | (__float_as_uint(lo) >> 16);
#endif
}

union FragBF { v16bf v; uint4 q[2]; };

// stage 16 f32 -> 16 bf16 (truncating) into LDS (two b128 writes)
DEVINL void stage_trunc(uint16_t* dst, const float* src) {
  const float4* s = (const float4*)src;
  const float4 f0 = s[0], f1 = s[1], f2 = s[2], f3 = s[3];
  uint4 o0, o1;
  o0.x = pack2_trunc(f0.x, f0.y); o0.y = pack2_trunc(f0.z, f0.w);
  o0.z = pack2_trunc(f1.x, f1.y); o0.w = pack2_trunc(f1.z, f1.w);
  o1.x = pack2_trunc(f2.x, f2.y); o1.y = pack2_trunc(f2.z, f2.w);
  o1.z = pack2_trunc(f3.x, f3.y); o1.w = pack2_trunc(f3.z, f3.w);
  ((uint4*)dst)[0] = o0;
  ((uint4*)dst)[1] = o1;
}

// stage 16 bf16 (raw copy) into LDS
DEVINL void stage_bf16(uint16_t* dst, const uint16_t* src, bool valid) {
  uint4 q0{}, q1{};
  if (valid) { q0 = ((const uint4*)src)[0]; q1 = ((const uint4*)src)[1]; }
  ((uint4*)dst)[0] = q0;
  ((uint4*)dst)[1] = q1;
}

// Load one 16x32 bf16 fragment (A or B) per the CDNA5 16-bit 16x32 layout:
// lane<16 -> K 0..7 and 16..23, lane>=16 -> K 8..15 and 24..31 (two ds_load_b128).
DEVINL v16bf load_frag(const uint16_t* tile, int row, int lane) {
  const uint16_t* p = tile + row * LDSS + ((lane >> 4) << 3);
  FragBF f;
  f.q[0] = ((const uint4*)p)[0];
  f.q[1] = ((const uint4*)(p + 16))[0];
  return f.v;
}

DEVINL v8f wmma_bf16(v16bf a, v16bf b, v8f c) {
  return __builtin_amdgcn_wmma_f32_16x16x32_bf16(false, a, false, b, (short)0, c,
                                                 false, false);
}

// ---------- kernel 0: x fp32 -> bf16 (RNE), done once off the hot loop ----------
__global__ __launch_bounds__(256) void convert_x(const float* __restrict__ x,
                                                 uint16_t* __restrict__ xbf) {
  const size_t i = ((size_t)blockIdx.x * 256 + threadIdx.x) * 8;
  const float4 f0 = *(const float4*)(x + i);
  const float4 f1 = *(const float4*)(x + i + 4);
  uint4 o;
  o.x = pack2_rne(f0.x, f0.y); o.y = pack2_rne(f0.z, f0.w);
  o.z = pack2_rne(f1.x, f1.y); o.w = pack2_rne(f1.z, f1.w);
  *(uint4*)(xbf + i) = o;
}

// ---------- kernel 1: routing ----------
__global__ __launch_bounds__(1024) void moe_route(
    const float* __restrict__ logits, int* __restrict__ pair_token,
    float* __restrict__ pair_gate, int* __restrict__ token_slot,
    int* __restrict__ tile_expert) {
  __shared__ int cnt[E_];
  __shared__ int off[E_ + 1];
  const int t = threadIdx.x;   // one token per thread, T_ == blockDim
  if (t < E_) cnt[t] = 0;
  for (int i = t; i < MAX_PAIRS; i += 1024) { pair_token[i] = -1; }
  if (t < MAX_MTILES) tile_expert[t] = -1;
  __syncthreads();

  float p[E_];
  float mx = -1e30f;
  #pragma unroll
  for (int e = 0; e < E_; ++e) { p[e] = logits[t * E_ + e]; mx = fmaxf(mx, p[e]); }
  #pragma unroll
  for (int e = 0; e < E_; ++e) p[e] = __expf(p[e] - mx);
  int i0 = 0;
  #pragma unroll
  for (int e = 1; e < E_; ++e) if (p[e] > p[i0]) i0 = e;
  int i1 = (i0 == 0) ? 1 : 0;
  #pragma unroll
  for (int e = 0; e < E_; ++e) if (e != i0 && p[e] > p[i1]) i1 = e;
  const float s2 = p[i0] + p[i1];
  const float w0 = p[i0] / s2, w1 = p[i1] / s2;

  const int pos0 = atomicAdd(&cnt[i0], 1);
  const int pos1 = atomicAdd(&cnt[i1], 1);
  __syncthreads();
  if (t == 0) {
    int acc = 0;
    for (int e = 0; e < E_; ++e) { off[e] = acc; acc += (cnt[e] + BM - 1) / BM * BM; }
    off[E_] = acc;
  }
  __syncthreads();
  if (t < E_) {
    for (int tile = off[t] / BM; tile < off[t + 1] / BM; ++tile) tile_expert[tile] = t;
  }
  const int s0 = off[i0] + pos0;
  const int s1 = off[i1] + pos1;
  pair_token[s0] = t; pair_gate[s0] = w0; token_slot[2 * t + 0] = s0;
  pair_token[s1] = t; pair_gate[s1] = w1; token_slot[2 * t + 1] = s1;
}

// ---------- kernel 2: gemm1 + fused bias + gated activation ----------
__global__ __launch_bounds__(128) void moe_gemm1(
    const uint16_t* __restrict__ xbf, const float* __restrict__ w13,
    const float* __restrict__ w13_bias, const int* __restrict__ pair_token,
    const int* __restrict__ tile_expert, uint16_t* __restrict__ act) {
  const int mt = blockIdx.y;
  const int e  = tile_expert[mt];
  if (e < 0) return;
  const int n0 = blockIdx.x * BN;   // glu column base in [0, I_)
  const int m0 = mt * BM;

  // double-buffered tiles
  __shared__ __align__(16) uint16_t As[2 * TILE_E];
  __shared__ __align__(16) uint16_t Bg[2 * TILE_E];
  __shared__ __align__(16) uint16_t Bl[2 * TILE_E];

  const int tid  = threadIdx.x;
  const int lane = tid & 31;
  const int wave = tid >> 5;
  const int wm = wave & 1, wn = wave >> 1;

  const int lrow  = tid >> 1;         // 0..63
  const int lhalf = (tid & 1) * 16;   // element offset within the 32-wide K tile

  const int  atok   = pair_token[m0 + lrow];
  const bool avalid = (atok >= 0);
  const uint16_t* aSrc = xbf + (size_t)(avalid ? atok : 0) * H_ + lhalf;
  const float* w13e = w13 + (size_t)e * (2 * I_) * H_;
  const float* gSrc = w13e + (size_t)(n0 + lrow) * H_ + lhalf;
  const float* lSrc = w13e + (size_t)(I_ + n0 + lrow) * H_ + lhalf;
  uint16_t* const aDst = As + lrow * LDSS + lhalf;
  uint16_t* const gDst = Bg + lrow * LDSS + lhalf;
  uint16_t* const lDst = Bl + lrow * LDSS + lhalf;

  const v8f vzero = {0.f, 0.f, 0.f, 0.f, 0.f, 0.f, 0.f, 0.f};
  v8f accG[2][2], accL[2][2];
  #pragma unroll
  for (int i = 0; i < 2; ++i)
    #pragma unroll
    for (int j = 0; j < 2; ++j) { accG[i][j] = vzero; accL[i][j] = vzero; }

  // prologue: stage k0 = 0 into buffer 0
  stage_bf16(aDst, aSrc, avalid);
  stage_trunc(gDst, gSrc);
  stage_trunc(lDst, lSrc);

  int buf = 0;
  for (int k0 = 0; k0 < H_; k0 += BK) {
    __syncthreads();
    const int nbuf = buf ^ 1;
    const int kn = k0 + BK;
    if (kn < H_) {   // stage next k-tile into the other buffer (overlaps WMMAs)
      stage_bf16(aDst + nbuf * TILE_E, aSrc + kn, avalid);
      stage_trunc(gDst + nbuf * TILE_E, gSrc + kn);
      stage_trunc(lDst + nbuf * TILE_E, lSrc + kn);
    }
    const uint16_t* A = As + buf * TILE_E;
    const uint16_t* G = Bg + buf * TILE_E;
    const uint16_t* L = Bl + buf * TILE_E;
    v16bf af[2], gf[2], lf[2];
    #pragma unroll
    for (int ms = 0; ms < 2; ++ms)
      af[ms] = load_frag(A, wm * 32 + ms * 16 + (lane & 15), lane);
    #pragma unroll
    for (int ns = 0; ns < 2; ++ns) {
      gf[ns] = load_frag(G, wn * 32 + ns * 16 + (lane & 15), lane);
      lf[ns] = load_frag(L, wn * 32 + ns * 16 + (lane & 15), lane);
    }
    #pragma unroll
    for (int ms = 0; ms < 2; ++ms)
      #pragma unroll
      for (int ns = 0; ns < 2; ++ns) {
        accG[ms][ns] = wmma_bf16(af[ms], gf[ns], accG[ms][ns]);
        accL[ms][ns] = wmma_bf16(af[ms], lf[ns], accL[ms][ns]);
      }
    buf = nbuf;
  }

  // Fused epilogue: h += bias; act = glu*sigmoid(1.702*glu)*(lin+1); store bf16.
  #pragma unroll
  for (int ns = 0; ns < 2; ++ns) {
    const int n = n0 + wn * 32 + ns * 16 + (lane & 15);
    const float bG = w13_bias[(size_t)e * (2 * I_) + n];
    const float bL = w13_bias[(size_t)e * (2 * I_) + I_ + n];
    #pragma unroll
    for (int ms = 0; ms < 2; ++ms) {
      const int mbase = m0 + wm * 32 + ms * 16 + ((lane >> 4) << 3);
      #pragma unroll
      for (int r = 0; r < 8; ++r) {
        const int mr = mbase + r;
        if (pair_token[mr] >= 0) {
          const float g = accG[ms][ns][r] + bG;
          const float l = accL[ms][ns][r] + bL;
          const float sg = 1.0f / (1.0f + __expf(-1.702f * g));
          const float a = g * sg * (l + 1.0f);
          act[(size_t)mr * I_ + n] = (uint16_t)bf16_bits_rne(a);
        }
      }
    }
  }
}

// ---------- kernel 3: gemm2 (+ w2 bias), per-pair y ----------
__global__ __launch_bounds__(128) void moe_gemm2(
    const uint16_t* __restrict__ act, const float* __restrict__ w2,
    const float* __restrict__ w2_bias, const int* __restrict__ pair_token,
    const int* __restrict__ tile_expert, float* __restrict__ y) {
  const int mt = blockIdx.y;
  const int e  = tile_expert[mt];
  if (e < 0) return;
  const int n0 = blockIdx.x * BN;   // hidden column base in [0, H_)
  const int m0 = mt * BM;

  __shared__ __align__(16) uint16_t As[2 * TILE_E];
  __shared__ __align__(16) uint16_t Bs[2 * TILE_E];

  const int tid  = threadIdx.x;
  const int lane = tid & 31;
  const int wave = tid >> 5;
  const int wm = wave & 1, wn = wave >> 1;

  const int lrow  = tid >> 1;
  const int lhalf = (tid & 1) * 16;

  const int  atok   = pair_token[m0 + lrow];
  const bool avalid = (atok >= 0);
  const uint16_t* aSrc = act + (size_t)(m0 + lrow) * I_ + lhalf;
  const float* w2e  = w2 + (size_t)e * H_ * I_;
  const float* bSrc = w2e + (size_t)(n0 + lrow) * I_ + lhalf;
  uint16_t* const aDst = As + lrow * LDSS + lhalf;
  uint16_t* const bDst = Bs + lrow * LDSS + lhalf;

  const v8f vzero = {0.f, 0.f, 0.f, 0.f, 0.f, 0.f, 0.f, 0.f};
  v8f acc[2][2];
  #pragma unroll
  for (int i = 0; i < 2; ++i)
    #pragma unroll
    for (int j = 0; j < 2; ++j) acc[i][j] = vzero;

  stage_bf16(aDst, aSrc, avalid);
  stage_trunc(bDst, bSrc);

  int buf = 0;
  for (int k0 = 0; k0 < I_; k0 += BK) {
    __syncthreads();
    const int nbuf = buf ^ 1;
    const int kn = k0 + BK;
    if (kn < I_) {
      stage_bf16(aDst + nbuf * TILE_E, aSrc + kn, avalid);
      stage_trunc(bDst + nbuf * TILE_E, bSrc + kn);
    }
    const uint16_t* A = As + buf * TILE_E;
    const uint16_t* B = Bs + buf * TILE_E;
    v16bf af[2], bfr[2];
    #pragma unroll
    for (int ms = 0; ms < 2; ++ms)
      af[ms] = load_frag(A, wm * 32 + ms * 16 + (lane & 15), lane);
    #pragma unroll
    for (int ns = 0; ns < 2; ++ns)
      bfr[ns] = load_frag(B, wn * 32 + ns * 16 + (lane & 15), lane);
    #pragma unroll
    for (int ms = 0; ms < 2; ++ms)
      #pragma unroll
      for (int ns = 0; ns < 2; ++ns)
        acc[ms][ns] = wmma_bf16(af[ms], bfr[ns], acc[ms][ns]);
    buf = nbuf;
  }

  #pragma unroll
  for (int ns = 0; ns < 2; ++ns) {
    const int n = n0 + wn * 32 + ns * 16 + (lane & 15);
    const float b = w2_bias[(size_t)e * H_ + n];
    #pragma unroll
    for (int ms = 0; ms < 2; ++ms) {
      const int mbase = m0 + wm * 32 + ms * 16 + ((lane >> 4) << 3);
      #pragma unroll
      for (int r = 0; r < 8; ++r) {
        const int mr = mbase + r;
        if (pair_token[mr] >= 0) y[(size_t)mr * H_ + n] = acc[ms][ns][r] + b;
      }
    }
  }
}

// ---------- kernel 4: gate-weighted combine ----------
__global__ __launch_bounds__(256) void moe_combine(
    const float* __restrict__ y, const int* __restrict__ token_slot,
    const float* __restrict__ pair_gate, float* __restrict__ out) {
  const int t  = blockIdx.x;
  const int s0 = token_slot[2 * t + 0];
  const int s1 = token_slot[2 * t + 1];
  const float g0 = pair_gate[s0];
  const float g1 = pair_gate[s1];
  const int c = threadIdx.x * 4;   // 256 threads * 4 = H_
  const float4 a = *(const float4*)(y + (size_t)s0 * H_ + c);
  const float4 b = *(const float4*)(y + (size_t)s1 * H_ + c);
  float4 o;
  o.x = g0 * a.x + g1 * b.x;
  o.y = g0 * a.y + g1 * b.y;
  o.z = g0 * a.z + g1 * b.z;
  o.w = g0 * a.w + g1 * b.w;
  *(float4*)(out + (size_t)t * H_ + c) = o;
}

// ---------- host launcher ----------
extern "C" void kernel_launch(void* const* d_in, const int* in_sizes, int n_in,
                              void* d_out, int out_size, void* d_ws, size_t ws_size,
                              hipStream_t stream) {
  const float* x        = (const float*)d_in[0];
  const float* logits   = (const float*)d_in[1];
  const float* w13      = (const float*)d_in[2];
  const float* w2       = (const float*)d_in[3];
  const float* w13_bias = (const float*)d_in[4];
  const float* w2_bias  = (const float*)d_in[5];
  // d_in[6] = top_k (fixed 2)

  char* ws = (char*)d_ws;
  size_t off = 0;
  auto alloc = [&](size_t bytes) -> void* {
    off = (off + 255) & ~(size_t)255;
    void* p = ws + off;
    off += bytes;
    return p;
  };
  int*      pair_token  = (int*)alloc((size_t)MAX_PAIRS * sizeof(int));
  float*    pair_gate   = (float*)alloc((size_t)MAX_PAIRS * sizeof(float));
  int*      token_slot  = (int*)alloc((size_t)T_ * 2 * sizeof(int));
  int*      tile_expert = (int*)alloc((size_t)MAX_MTILES * sizeof(int));
  uint16_t* xbf         = (uint16_t*)alloc((size_t)T_ * H_ * sizeof(uint16_t));
  uint16_t* act         = (uint16_t*)alloc((size_t)MAX_PAIRS * I_ * sizeof(uint16_t));
  float*    yb          = (float*)alloc((size_t)MAX_PAIRS * H_ * sizeof(float));
  (void)ws_size; (void)in_sizes; (void)n_in; (void)out_size;

  convert_x<<<(T_ * H_) / (256 * 8), 256, 0, stream>>>(x, xbf);
  moe_route<<<1, 1024, 0, stream>>>(logits, pair_token, pair_gate, token_slot,
                                    tile_expert);
  moe_gemm1<<<dim3(I_ / BN, MAX_MTILES), 128, 0, stream>>>(
      xbf, w13, w13_bias, pair_token, tile_expert, act);
  moe_gemm2<<<dim3(H_ / BN, MAX_MTILES), 128, 0, stream>>>(
      act, w2, w2_bias, pair_token, tile_expert, yb);
  moe_combine<<<T_, 256, 0, stream>>>(yb, token_slot, pair_gate, (float*)d_out);
}
